// NonLocalBlock_22110491639889
// MI455X (gfx1250) — matrix operand order repaired and checked by reading, not compiled, not run on hardware
//
#include <hip/hip_runtime.h>
#include <hip/hip_bf16.h>

// ---------------------------------------------------------------------------
// Non-Local block (self-attention) for MI455X / gfx1250, wave32 + WMMA f16
// + async global->LDS staging (ASYNCcnt).  B=8, C=512, CI=256, N=3136.
// ---------------------------------------------------------------------------

typedef _Float16 half_t;
typedef __attribute__((ext_vector_type(16))) _Float16 v16h;
typedef __attribute__((ext_vector_type(8)))  float    v8f;
typedef __attribute__((ext_vector_type(4)))  int      v4i_t;

static constexpr int B_  = 8;
static constexpr int C_  = 512;
static constexpr int CI_ = 256;
static constexpr int N_  = 3136;   // 56*56
static constexpr float EPS_ = 1e-5f;

// workspace layout (bytes)
static constexpr size_t WCAT_OFF  = 0;                       // 768*512 f16
static constexpr size_t WW_OFF    = WCAT_OFF  + 768*512*2;   // 512*256 f16
static constexpr size_t THETA_OFF = WW_OFF    + 512*256*2;   // [B][N][CI] f16
static constexpr size_t PHI_OFF   = THETA_OFF + (size_t)B_*N_*CI_*2;
static constexpr size_t G_OFF     = PHI_OFF   + (size_t)B_*N_*CI_*2; // [B][CI][N] f16
static constexpr size_t Y_OFF     = G_OFF     + (size_t)B_*N_*CI_*2; // [B][N][CI] f16
static constexpr size_t SUM_OFF   = Y_OFF     + (size_t)B_*N_*CI_*2; // 512 f32
static constexpr size_t SUMSQ_OFF = SUM_OFF   + 512*4;               // 512 f32
static constexpr size_t XT_OFF    = SUMSQ_OFF + 512*4;       // [B][N][C] f16

// ---------------------------------------------------------------------------
// gfx1250 async global->LDS copies (ASYNCcnt).  Guarded: falls back to
// synchronous copies if the toolchain lacks the builtins.
// ---------------------------------------------------------------------------
#if defined(__has_builtin)
#  if __has_builtin(__builtin_amdgcn_global_load_async_to_lds_b128) && \
      __has_builtin(__builtin_amdgcn_s_wait_asynccnt)
#    define HAVE_ASYNC_LDS 1
#  endif
#endif
#ifndef HAVE_ASYNC_LDS
#  define HAVE_ASYNC_LDS 0
#endif

__device__ __forceinline__ void async_copy_b128(half_t* dst_lds, const half_t* src) {
#if HAVE_ASYNC_LDS
  __builtin_amdgcn_global_load_async_to_lds_b128(
      (__attribute__((address_space(1))) v4i_t*)(src),
      (__attribute__((address_space(3))) v4i_t*)(dst_lds),
      0, 0);
#else
  *(uint4*)dst_lds = *(const uint4*)src;
#endif
}

__device__ __forceinline__ void wait_async_lds() {
#if HAVE_ASYNC_LDS
  __builtin_amdgcn_s_wait_asynccnt(0);
#endif
}

// ---------------------------------------------------------------------------
// WMMA fragment helpers (layouts per CDNA5 ISA 7.12.2, wave32)
// ---------------------------------------------------------------------------
__device__ __forceinline__ v8f wmma_f16(v16h a, v16h b, v8f c) {
  return __builtin_amdgcn_wmma_f32_16x16x32_f16(false, a, false, b, (short)0, c,
                                                false, false);
}

// A fragment (16x32 f16), source row-major: element(m,k) at p[m*ld + k].
__device__ __forceinline__ v16h load_a_tile(const half_t* p, int ld) {
  const int lane = threadIdx.x & 31;
  const half_t* r = p + (lane & 15) * ld + (lane >> 4) * 8;
  union { v16h v; uint4 q[2]; } u;
  u.q[0] = *(const uint4*)(r);
  u.q[1] = *(const uint4*)(r + 16);
  return u.v;
}

// B fragment (32x16 f16), K-contiguous source: element(k,j) at p[j*ld + k].
__device__ __forceinline__ v16h load_b_tile(const half_t* p, int ld) {
  const int lane = threadIdx.x & 31;
  const half_t* r = p + (lane & 15) * ld + (lane >> 4) * 8;
  union { v16h v; uint4 q[2]; } u;
  u.q[0] = *(const uint4*)(r);
  u.q[1] = *(const uint4*)(r + 16);
  return u.v;
}

// C/D fragment (16x16 f32): vgpr r, lane L -> row = r + 8*(L>>4), col = L&15.

// ---------------------------------------------------------------------------
// K0a: convert weights to f16.  Wcat = [theta_w; phi_w; g_w]  (768 x 512)
// ---------------------------------------------------------------------------
__global__ void convw_kernel(const float* __restrict__ tw, const float* __restrict__ pw,
                             const float* __restrict__ gw, const float* __restrict__ ww,
                             half_t* __restrict__ Wcat, half_t* __restrict__ Wf16) {
  int i = blockIdx.x * 256 + threadIdx.x;
  if (i < 768 * 512) {
    int r = i >> 9, c = i & 511;
    float v = (r < 256) ? tw[r * 512 + c]
            : (r < 512) ? pw[(r - 256) * 512 + c]
                        : gw[(r - 512) * 512 + c];
    Wcat[i] = (half_t)v;
  }
  if (i < 512 * 256) Wf16[i] = (half_t)ww[i];
}

// ---------------------------------------------------------------------------
// K0b: x [b][c][n] f32 -> xT [b][n][c] f16 (tiled transpose through LDS).
// ---------------------------------------------------------------------------
__global__ __launch_bounds__(256) void convx_kernel(const float* __restrict__ x,
                                                    half_t* __restrict__ xT) {
  __shared__ half_t tile[32][33];
  const int b  = blockIdx.z;
  const int c0 = blockIdx.y * 32;
  const int nn0 = blockIdx.x * 32;
  const int tc  = threadIdx.x & 31;
  const int tr4 = threadIdx.x >> 5;   // 0..7, 4 rows each

  const float* xb = x + ((size_t)b * C_ + c0) * N_ + nn0;
  #pragma unroll
  for (int i = 0; i < 4; ++i) {
    int r = tr4 * 4 + i;
    tile[r][tc] = (half_t)xb[(size_t)r * N_ + tc];
  }
  __syncthreads();
  half_t* xo = xT + ((size_t)b * N_ + nn0) * C_ + c0;
  #pragma unroll
  for (int i = 0; i < 4; ++i) {
    int r = tr4 * 4 + i;
    xo[(size_t)r * C_ + tc] = tile[tc][r];
  }
}

__global__ void zero_kernel(float* __restrict__ p, int n) {
  int i = blockIdx.x * blockDim.x + threadIdx.x;
  if (i < n) p[i] = 0.f;
}

// ---------------------------------------------------------------------------
// K1: fused projections  tpg = Wcat(768x512) @ xT^T per batch.
// Pure register-tile GEMM: no LDS, no barriers.
// ---------------------------------------------------------------------------
__global__ __launch_bounds__(128) void proj_kernel(
    const half_t* __restrict__ xT, const half_t* __restrict__ Wcat,
    const float* __restrict__ tb, const float* __restrict__ pb,
    const float* __restrict__ gb,
    half_t* __restrict__ thetaT, half_t* __restrict__ phiT, half_t* __restrict__ gN) {
  const int lane = threadIdx.x & 31;
  const int wv   = threadIdx.x >> 5;
  const int grp  = lane >> 4;
  const int lj   = lane & 15;

  const int b  = blockIdx.z;
  const int n0 = blockIdx.y * 16;
  const int o0 = (blockIdx.x * 4 + wv) * 16;

  const half_t* xb = xT + ((size_t)b * N_ + n0) * C_;
  v8f acc = {};
  #pragma unroll
  for (int ks = 0; ks < C_ / 32; ++ks) {
    v16h a  = load_a_tile(Wcat + (size_t)o0 * C_ + ks * 32, C_);
    v16h bb = load_b_tile(xb + ks * 32, C_);
    acc = wmma_f16(a, bb, acc);
  }

  const int n = n0 + lj;
  #pragma unroll
  for (int r = 0; r < 8; ++r) {
    const int orow = o0 + r + 8 * grp;
    float bias = (orow < 256) ? tb[orow] : (orow < 512) ? pb[orow - 256] : gb[orow - 512];
    float val = acc[r] + bias;
    if (orow < 256)
      thetaT[((size_t)b * N_ + n) * CI_ + orow] = (half_t)val;
    else if (orow < 512)
      phiT[((size_t)b * N_ + n) * CI_ + (orow - 256)] = (half_t)val;
    else
      gN[((size_t)b * CI_ + (orow - 512)) * N_ + n] = (half_t)val;
  }
}

// ---------------------------------------------------------------------------
// K2: flash attention with double-buffered async K/V staging.
// ---------------------------------------------------------------------------
__device__ __forceinline__ void stage_tiles(const half_t* __restrict__ phib,
                                            const half_t* __restrict__ gb_,
                                            int m0,
                                            half_t (*sKb)[256], half_t (*sVb)[32]) {
  half_t* kdst = &sKb[0][0];
  half_t* vdst = &sVb[0][0];
  // K tile: 32 keys x 128 ci  (512 x b128)
  #pragma unroll
  for (int i = 0; i < 4; ++i) {
    int idx = i * 128 + threadIdx.x;
    int row = idx >> 4, col = idx & 15;
    async_copy_b128(kdst + idx * 8, phib + (size_t)(m0 + row) * CI_ + col * 8);
  }
  // V tile: 256 ci x 32 keys  (1024 x b128)
  #pragma unroll
  for (int i = 0; i < 8; ++i) {
    int idx = i * 128 + threadIdx.x;
    int row = idx >> 2, col = idx & 3;
    async_copy_b128(vdst + idx * 8, gb_ + (size_t)row * N_ + m0 + col * 8);
  }
}

__global__ __launch_bounds__(128) void attn_kernel(
    const half_t* __restrict__ thetaT, const half_t* __restrict__ phiT,
    const half_t* __restrict__ gN, half_t* __restrict__ yT) {
  const int lane = threadIdx.x & 31;
  const int wv   = threadIdx.x >> 5;
  const int grp  = lane >> 4;
  const int lj   = lane & 15;

  const int b  = blockIdx.y;
  const int q0 = blockIdx.x * 64 + wv * 16;

  __shared__ half_t sK[2][32][256];   // 32 KB double-buffered K
  __shared__ half_t sV[2][256][32];   // 32 KB double-buffered V
  __shared__ half_t sP[4][16][32];    //  4 KB per-wave softmax scratch

  // Q fragments in registers (8 frags x 8 VGPRs)
  v16h qf[8];
  #pragma unroll
  for (int t = 0; t < 8; ++t)
    qf[t] = load_a_tile(thetaT + ((size_t)b * N_ + q0) * CI_ + t * 32, CI_);

  v8f yacc[16];
  #pragma unroll
  for (int c = 0; c < 16; ++c) yacc[c] = (v8f){};
  float m_run[8], l_run[8];
  #pragma unroll
  for (int r = 0; r < 8; ++r) { m_run[r] = -1e30f; l_run[r] = 0.f; }

  const half_t* phib = phiT + (size_t)b * N_ * CI_;
  const half_t* gb_  = gN   + (size_t)b * CI_ * N_;

  int cur = 0;
  stage_tiles(phib, gb_, 0, sK[0], sV[0]);

  for (int kt = 0; kt < N_ / 32; ++kt) {
    wait_async_lds();          // this wave's async copies landed
    __syncthreads();           // all waves' copies published
    if (kt + 1 < N_ / 32)      // prefetch next tile into the other buffer
      stage_tiles(phib, gb_, (kt + 1) * 32, sK[cur ^ 1], sV[cur ^ 1]);

    // S = Q @ K^T for two 16-key sub-tiles
    v8f s0 = (v8f){}, s1 = (v8f){};
    #pragma unroll
    for (int t = 0; t < 8; ++t) {
      v16h b0 = load_b_tile(&sK[cur][0][0]  + t * 32, 256);
      v16h b1 = load_b_tile(&sK[cur][16][0] + t * 32, 256);
      s0 = wmma_f16(qf[t], b0, s0);
      s1 = wmma_f16(qf[t], b1, s1);
    }

    // online softmax (row stats live in 16-lane groups == C-frag row mapping)
    float sc[8];
    #pragma unroll
    for (int r = 0; r < 8; ++r) {
      float a0 = s0[r], a1 = s1[r];
      float mx = fmaxf(a0, a1);
      mx = fmaxf(mx, __shfl_xor(mx, 1));
      mx = fmaxf(mx, __shfl_xor(mx, 2));
      mx = fmaxf(mx, __shfl_xor(mx, 4));
      mx = fmaxf(mx, __shfl_xor(mx, 8));
      float mnew = fmaxf(m_run[r], mx);
      float corr = __expf(m_run[r] - mnew);
      float p0 = __expf(a0 - mnew);
      float p1 = __expf(a1 - mnew);
      float rs = p0 + p1;
      rs += __shfl_xor(rs, 1);
      rs += __shfl_xor(rs, 2);
      rs += __shfl_xor(rs, 4);
      rs += __shfl_xor(rs, 8);
      l_run[r] = l_run[r] * corr + rs;
      m_run[r] = mnew;
      sc[r] = corr;
      sP[wv][r + 8 * grp][lj]      = (half_t)p0;
      sP[wv][r + 8 * grp][16 + lj] = (half_t)p1;
    }
    #pragma unroll
    for (int c = 0; c < 16; ++c)
      #pragma unroll
      for (int r = 0; r < 8; ++r) yacc[c][r] *= sc[r];

    // y += P @ V  (P re-laid out through LDS as an A fragment)
    v16h pa = load_a_tile(&sP[wv][0][0], 32);
    #pragma unroll
    for (int c = 0; c < 16; ++c) {
      v16h vb = load_b_tile(&sV[cur][c * 16][0], 32);
      yacc[c] = wmma_f16(pa, vb, yacc[c]);
    }
    cur ^= 1;
  }

  // normalize and store yT [b][n][ci] f16
  float rl[8];
  #pragma unroll
  for (int r = 0; r < 8; ++r) rl[r] = 1.0f / l_run[r];
  #pragma unroll
  for (int c = 0; c < 16; ++c) {
    #pragma unroll
    for (int r = 0; r < 8; ++r) {
      const int q  = q0 + r + 8 * grp;
      const int ci = c * 16 + lj;
      yT[((size_t)b * N_ + q) * CI_ + ci] = (half_t)(yacc[c][r] * rl[r]);
    }
  }
}

// ---------------------------------------------------------------------------
// K3a: w_y = W(512x256) @ y ; accumulate per-channel sum / sumsq for BN.
// ---------------------------------------------------------------------------
__global__ __launch_bounds__(128) void wsum_kernel(
    const half_t* __restrict__ Wf16, const half_t* __restrict__ yT,
    const float* __restrict__ wb, float* __restrict__ gsum, float* __restrict__ gsq) {
  const int lane = threadIdx.x & 31;
  const int wv   = threadIdx.x >> 5;
  const int grp  = lane >> 4;
  const int lj   = lane & 15;

  const int o0 = blockIdx.x * 16;
  const int b  = blockIdx.z;

  __shared__ float sSum[4][16];
  __shared__ float sSq[4][16];

  float psum[8], psq[8];
  #pragma unroll
  for (int r = 0; r < 8; ++r) { psum[r] = 0.f; psq[r] = 0.f; }

  const half_t* yb = yT + (size_t)b * N_ * CI_;
  for (int it = 0; it < 7; ++it) {
    const int n0 = (blockIdx.y * 28 + wv * 7 + it) * 16;
    v8f acc = (v8f){};
    #pragma unroll
    for (int t = 0; t < 8; ++t) {
      v16h a  = load_a_tile(Wf16 + (size_t)o0 * CI_ + t * 32, CI_);
      v16h bb = load_b_tile(yb + (size_t)n0 * CI_ + t * 32, CI_);
      acc = wmma_f16(a, bb, acc);
    }
    #pragma unroll
    for (int r = 0; r < 8; ++r) {
      float v = acc[r] + wb[o0 + r + 8 * grp];
      float s = v, q = v * v;
      s += __shfl_xor(s, 1); q += __shfl_xor(q, 1);
      s += __shfl_xor(s, 2); q += __shfl_xor(q, 2);
      s += __shfl_xor(s, 4); q += __shfl_xor(q, 4);
      s += __shfl_xor(s, 8); q += __shfl_xor(q, 8);
      psum[r] += s; psq[r] += q;
    }
  }
  if (lj == 0) {
    #pragma unroll
    for (int r = 0; r < 8; ++r) {
      sSum[wv][r + 8 * grp] = psum[r];
      sSq[wv][r + 8 * grp]  = psq[r];
    }
  }
  __syncthreads();
  if (threadIdx.x < 16) {
    float s = 0.f, q = 0.f;
    for (int w = 0; w < 4; ++w) { s += sSum[w][threadIdx.x]; q += sSq[w][threadIdx.x]; }
    atomicAdd(&gsum[o0 + threadIdx.x], s);
    atomicAdd(&gsq[o0 + threadIdx.x], q);
  }
}

// ---------------------------------------------------------------------------
// K3b: recompute w_y tile, fuse BatchNorm (batch stats) + residual, store f32.
// ---------------------------------------------------------------------------
__global__ __launch_bounds__(128) void final_kernel(
    const half_t* __restrict__ Wf16, const half_t* __restrict__ yT,
    const float* __restrict__ wb, const float* __restrict__ gsum,
    const float* __restrict__ gsq, const float* __restrict__ gamma,
    const float* __restrict__ beta, const float* __restrict__ x,
    float* __restrict__ out) {
  const int lane = threadIdx.x & 31;
  const int wv   = threadIdx.x >> 5;
  const int grp  = lane >> 4;
  const int lj   = lane & 15;

  const int o0 = blockIdx.x * 16;
  const int b  = blockIdx.z;
  const int n0 = (blockIdx.y * 4 + wv) * 16;

  const half_t* yb = yT + (size_t)b * N_ * CI_;
  v8f acc = (v8f){};
  #pragma unroll
  for (int t = 0; t < 8; ++t) {
    v16h a  = load_a_tile(Wf16 + (size_t)o0 * CI_ + t * 32, CI_);
    v16h bb = load_b_tile(yb + (size_t)n0 * CI_ + t * 32, CI_);
    acc = wmma_f16(a, bb, acc);
  }

  const float cnt = (float)B_ * (float)N_;
  const int n = n0 + lj;
  #pragma unroll
  for (int r = 0; r < 8; ++r) {
    const int o = o0 + r + 8 * grp;
    float mean = gsum[o] / cnt;
    float var  = gsq[o] / cnt - mean * mean;
    float v = acc[r] + wb[o];
    v = (v - mean) * rsqrtf(var + EPS_) * gamma[o] + beta[o]
        + x[((size_t)b * C_ + o) * N_ + n];
    out[((size_t)b * C_ + o) * N_ + n] = v;
  }
}

// ---------------------------------------------------------------------------
extern "C" void kernel_launch(void* const* d_in, const int* in_sizes, int n_in,
                              void* d_out, int out_size, void* d_ws, size_t ws_size,
                              hipStream_t stream) {
  const float* x       = (const float*)d_in[0];
  const float* theta_w = (const float*)d_in[1];
  const float* theta_b = (const float*)d_in[2];
  const float* phi_w   = (const float*)d_in[3];
  const float* phi_b   = (const float*)d_in[4];
  const float* g_w     = (const float*)d_in[5];
  const float* g_b     = (const float*)d_in[6];
  const float* w_w     = (const float*)d_in[7];
  const float* w_b     = (const float*)d_in[8];
  const float* gamma   = (const float*)d_in[9];
  const float* beta    = (const float*)d_in[10];
  float* out = (float*)d_out;

  char* ws = (char*)d_ws;
  half_t* Wcat   = (half_t*)(ws + WCAT_OFF);
  half_t* Wf16   = (half_t*)(ws + WW_OFF);
  half_t* thetaT = (half_t*)(ws + THETA_OFF);
  half_t* phiT   = (half_t*)(ws + PHI_OFF);
  half_t* gN     = (half_t*)(ws + G_OFF);
  half_t* yT     = (half_t*)(ws + Y_OFF);
  float*  gsum   = (float*)(ws + SUM_OFF);
  float*  gsq    = (float*)(ws + SUMSQ_OFF);
  half_t* xT     = (half_t*)(ws + XT_OFF);

  // K0a: weights -> f16
  convw_kernel<<<dim3((768 * 512 + 255) / 256), dim3(256), 0, stream>>>(
      theta_w, phi_w, g_w, w_w, Wcat, Wf16);

  // K0b: x -> xT f16 (transpose)
  convx_kernel<<<dim3(N_ / 32, C_ / 32, B_), dim3(256), 0, stream>>>(x, xT);

  // K1: projections (LDS-free register-tile GEMM)
  proj_kernel<<<dim3(12, N_ / 16, B_), dim3(128), 0, stream>>>(
      xT, Wcat, theta_b, phi_b, g_b, thetaT, phiT, gN);

  // K2: flash attention with async double-buffered K/V staging
  attn_kernel<<<dim3(N_ / 64, B_), dim3(128), 0, stream>>>(thetaT, phiT, gN, yT);

  // zero BN accumulators
  zero_kernel<<<dim3(4), dim3(256), 0, stream>>>(gsum, 1024);

  // K3a: BN statistics
  wsum_kernel<<<dim3(C_ / 16, 7, B_), dim3(128), 0, stream>>>(Wf16, yT, w_b, gsum, gsq);

  // K3b: final projection + BN + residual
  final_kernel<<<dim3(C_ / 16, 49, B_), dim3(128), 0, stream>>>(
      Wf16, yT, w_b, gsum, gsq, gamma, beta, x, out);
}
